// CausalAttention_17497696764057
// MI455X (gfx1250) — matrix operand order repaired and checked by reading, not compiled, not run on hardware
//
#include <hip/hip_runtime.h>
#include <hip/hip_bf16.h>

// ---------------------------------------------------------------------------
// CausalAttention (GQA + RoPE) for gfx1250 / MI455X.
// All matrix math via v_wmma_f32_16x16x32_bf16 (wave32 WMMA, f32 accumulate).
// Fragment traffic uses vectorized ds_load_b128 (2 per fragment).
// ---------------------------------------------------------------------------

typedef __attribute__((ext_vector_type(16))) __bf16 v16bf;
typedef __attribute__((ext_vector_type(8)))  __bf16 v8bf;
typedef __attribute__((ext_vector_type(2)))  __bf16 v2bf;
typedef __attribute__((ext_vector_type(8)))  float  v8f;

static constexpr int Bb    = 2;
static constexpr int Tt    = 2048;
static constexpr int DIM   = 1024;
static constexpr int HEADS = 16;
static constexpr int KVH   = 4;
static constexpr int HD    = 64;
static constexpr int KVD   = KVH * HD;   // 256

__device__ __forceinline__ v8f wmma_bf16(v16bf a, v16bf b, v8f c) {
  return __builtin_amdgcn_wmma_f32_16x16x32_bf16(
      /*neg_a=*/false, a, /*neg_b=*/false, b,
      /*c_mod=*/(short)0, c, /*reuse_a=*/false, /*reuse_b=*/false);
}

// A fragment (16x32, M x K) from a row-major tile, leading dim ldm (elems).
// ISA 7.12.2: lane&15 = M; per lane the 16 values are two contiguous
// 8-element runs: K = half*8 + [0,8) and K = 16 + half*8 + [0,8).
// -> two ds_load_b128 per fragment (ldm*2 and chunk offsets 16B-aligned).
__device__ __forceinline__ v16bf load_frag_a(const __bf16* base, int ldm) {
  const int lane = threadIdx.x & 31;
  const __bf16* rp = base + (lane & 15) * ldm + ((lane >> 4) << 3);
  v8bf lo = *(const v8bf*)(rp);
  v8bf hi = *(const v8bf*)(rp + 16);
  return __builtin_shufflevector(lo, hi, 0, 1, 2, 3, 4, 5, 6, 7,
                                 8, 9, 10, 11, 12, 13, 14, 15);
}

// B fragment (32x16, K x N) from an N-MAJOR tile: base points at [n=0][k=0],
// leading dim ldm (elems of K per N row).
// ISA layout: lane&15 = N; per lane the 16 values are two 8-element runs:
// K = half*8 + [0,8) and K = 16 + half*8 + [0,8) -> two ds_load_b128.
__device__ __forceinline__ v16bf load_frag_bT(const __bf16* base, int ldm) {
  const int lane = threadIdx.x & 31;
  const __bf16* rp = base + (lane & 15) * ldm + ((lane >> 4) << 3);
  v8bf lo = *(const v8bf*)(rp);
  v8bf hi = *(const v8bf*)(rp + 16);
  return __builtin_shufflevector(lo, hi, 0, 1, 2, 3, 4, 5, 6, 7,
                                 8, 9, 10, 11, 12, 13, 14, 15);
}

// ---------------------------------------------------------------------------
// GEMM: C[M,N] = A[M,K] * Bw[K,N].  A is f32 or bf16 (template), Bw f32.
// Block 256 threads (8 waves), tile 128x64, k-step 64 (2 WMMA K chunks).
// Wave grid 4x2; each wave: 32x32 sub-tile = 2x2 C frags, 8 WMMA / k-step.
// W tile is stored N-major in LDS so B fragments are contiguous loads.
// ---------------------------------------------------------------------------
template <bool ABF16>
__global__ __launch_bounds__(256) void gemm_wmma_k(
    const void* __restrict__ Ap, const float* __restrict__ Bw,
    float* __restrict__ C, int M, int N, int K) {
  __shared__ __bf16 lds_a[128][72];    // 128 x 64 (row-major), +8 pad
  __shared__ __bf16 lds_bt[64][72];    // 64 n x 64 k (N-major), +8 pad

  const int tid  = threadIdx.x;
  const int lane = tid & 31;
  const int wid  = tid >> 5;
  const int wm   = wid >> 1;           // 0..3
  const int wn   = wid & 1;            // 0..1
  const int m0   = blockIdx.y * 128;
  const int n0   = blockIdx.x * 64;

  const float*  Af = (const float*)Ap;
  const __bf16* Ab = (const __bf16*)Ap;

  v8f c[2][2];
  v8f zero = {};
  c[0][0] = zero; c[0][1] = zero; c[1][0] = zero; c[1][1] = zero;

  for (int k0 = 0; k0 < K; k0 += 64) {
    // ---- stage A tile: 128x64, 32 contiguous elems/thread, v8bf stores ----
    {
      const int arow = tid >> 1;
      const int acol = (tid & 1) * 32;
      const size_t g = (size_t)(m0 + arow) * K + k0 + acol;
#pragma unroll
      for (int v = 0; v < 4; ++v) {
        v8bf pk;
#pragma unroll
        for (int e = 0; e < 8; ++e)
          pk[e] = ABF16 ? Ab[g + v * 8 + e] : (__bf16)Af[g + v * 8 + e];
        *(v8bf*)&lds_a[arow][acol + v * 8] = pk;
      }
    }
    // ---- stage B tile transposed: read K-row (coalesced), scatter N-major ----
    {
      const int kr = tid >> 2;
      const int nc = (tid & 3) * 16;
      const size_t g = (size_t)(k0 + kr) * N + n0 + nc;
#pragma unroll
      for (int e = 0; e < 16; ++e)
        lds_bt[nc + e][kr] = (__bf16)Bw[g + e];
    }
    if (k0 + 64 < K) {  // next B slab -> global_prefetch_b8
      __builtin_prefetch(&Bw[(size_t)(k0 + 64 + (tid >> 2)) * N + n0], 0, 1);
    }
    __syncthreads();

#pragma unroll
    for (int kc = 0; kc < 64; kc += 32) {
      v16bf a0 = load_frag_a(&lds_a[wm * 32][kc],       72);
      v16bf a1 = load_frag_a(&lds_a[wm * 32 + 16][kc],  72);
      v16bf b0 = load_frag_bT(&lds_bt[wn * 32][kc],     72);
      v16bf b1 = load_frag_bT(&lds_bt[wn * 32 + 16][kc], 72);
      c[0][0] = wmma_bf16(a0, b0, c[0][0]);
      c[0][1] = wmma_bf16(a0, b1, c[0][1]);
      c[1][0] = wmma_bf16(a1, b0, c[1][0]);
      c[1][1] = wmma_bf16(a1, b1, c[1][1]);
    }
    __syncthreads();
  }

  const int half = lane >> 4, nn = lane & 15;
#pragma unroll
  for (int i = 0; i < 2; ++i)
#pragma unroll
    for (int j = 0; j < 2; ++j)
#pragma unroll
      for (int r = 0; r < 8; ++r) {
        const int row = m0 + wm * 32 + i * 16 + r + 8 * half;
        const int col = n0 + wn * 32 + j * 16 + nn;
        C[(size_t)row * N + col] = c[i][j][r];
      }
}

// ---------------------------------------------------------------------------
// Flash attention. grid = (T/64, HEADS, B), block = 128 (4 waves).
// Wave w owns query rows [q0 + 16w, q0 + 16w + 16). Streams 32-key tiles,
// online softmax, WMMA for Q·K^T and P·V. Writes bf16 [B,T,H,HD] output.
// K kept [key][hd] (already N-major for K^T); V transposed to [hd][key].
// ---------------------------------------------------------------------------
__global__ __launch_bounds__(128) void flash_attn_k(
    const float* __restrict__ Qf, const float* __restrict__ Kf,
    const float* __restrict__ Vf, const float* __restrict__ sinp,
    const float* __restrict__ cosp, const float* __restrict__ maskp,
    __bf16* __restrict__ Ob) {
  const int q0  = blockIdx.x * 64;
  const int h   = blockIdx.y;
  const int b   = blockIdx.z;
  const int hkv = h >> 2;             // GROUPS = HEADS/KVH = 4
  const int tid = threadIdx.x;
  const int lane = tid & 31;
  const int w    = tid >> 5;
  const int half = lane >> 4;
  const int nn   = lane & 15;

  __shared__ __bf16 q_s[64][72];      // queries row-major [q][hd]
  __shared__ __bf16 k_s[32][72];      // keys    row-major [key][hd] (N-major for K^T)
  __shared__ __bf16 v_t[64][40];      // values  N-major   [hd][key]
  __shared__ __bf16 p_s[4][16][40];   // per-wave P bounce [q][key]
  __shared__ float  mk_s[32];

  // ---- stage Q tile with RoPE (64 rows x 32 pairs; 16 pairs/thread) ----
#pragma unroll
  for (int e = 0; e < 16; ++e) {
    const int p   = tid * 16 + e;
    const int row = p >> 5;
    const int d   = (p & 31) << 1;
    const int pos = q0 + row;
    const size_t g = (size_t)(b * Tt + pos) * DIM + h * HD + d;
    const float x0 = Qf[g], x1 = Qf[g + 1];
    const float s0 = sinp[pos * HD + d],     c0 = cosp[pos * HD + d];
    const float s1 = sinp[pos * HD + d + 1], c1 = cosp[pos * HD + d + 1];
    v2bf pr;
    pr[0] = (__bf16)(x0 * c0 - x1 * s0);
    pr[1] = (__bf16)(x1 * c1 + x0 * s1);
    *(v2bf*)&q_s[row][d] = pr;
  }

  float m_i[8], l_i[8];
  v8f o[4];
  v8f zero = {};
#pragma unroll
  for (int j = 0; j < 4; ++j) o[j] = zero;
#pragma unroll
  for (int r = 0; r < 8; ++r) { m_i[r] = -1e30f; l_i[r] = 0.0f; }

  const int ntiles = q0 / 32 + 2;     // causal: keys up to q0+63
  for (int t = 0; t < ntiles; ++t) {
    const int k0 = t * 32;
    // ---- stage K tile with RoPE (32 rows x 32 pairs; 8 pairs/thread) ----
#pragma unroll
    for (int e = 0; e < 8; ++e) {
      const int p   = tid * 8 + e;
      const int row = p >> 5;
      const int d   = (p & 31) << 1;
      const int pos = k0 + row;
      const size_t g = (size_t)(b * Tt + pos) * KVD + hkv * HD + d;
      const float x0 = Kf[g], x1 = Kf[g + 1];
      const float s0 = sinp[pos * HD + d],     c0 = cosp[pos * HD + d];
      const float s1 = sinp[pos * HD + d + 1], c1 = cosp[pos * HD + d + 1];
      v2bf pr;
      pr[0] = (__bf16)(x0 * c0 - x1 * s0);
      pr[1] = (__bf16)(x1 * c1 + x0 * s1);
      *(v2bf*)&k_s[row][d] = pr;
    }
    // ---- stage V tile transposed: coalesced read [key][hd], scatter [hd][key] ----
#pragma unroll
    for (int e = 0; e < 16; ++e) {
      const int idx = tid * 16 + e;
      const int key = idx >> 6;
      const int d   = idx & 63;
      v_t[d][key] = (__bf16)Vf[(size_t)(b * Tt + k0 + key) * KVD + hkv * HD + d];
    }
    if (tid < 32) mk_s[tid] = maskp[b * Tt + k0 + tid];
    __syncthreads();

    // ---- scores: S(16x32) = Q(16x64) · K^T(64x32), two K=32 chunks ----
    v8f s0 = zero, s1 = zero;
    {
      v16bf aq0 = load_frag_a(&q_s[w * 16][0],  72);
      v16bf aq1 = load_frag_a(&q_s[w * 16][32], 72);
      v16bf bk;
      bk = load_frag_bT(&k_s[0][0],   72); s0 = wmma_bf16(aq0, bk, s0);
      bk = load_frag_bT(&k_s[0][32],  72); s0 = wmma_bf16(aq1, bk, s0);
      bk = load_frag_bT(&k_s[16][0],  72); s1 = wmma_bf16(aq0, bk, s1);
      bk = load_frag_bT(&k_s[16][32], 72); s1 = wmma_bf16(aq1, bk, s1);
    }

    // ---- online softmax (rows live in 16-lane groups; reduce via shfl) ----
    float corr[8];
#pragma unroll
    for (int r = 0; r < 8; ++r) {
      const int qrow = q0 + w * 16 + r + 8 * half;
      float v0 = s0[r] * 0.125f;   // 1/sqrt(HD)
      float v1 = s1[r] * 0.125f;
      if ((k0 + nn) > qrow      || mk_s[nn]      <= 0.0f) v0 = -1e30f;
      if ((k0 + 16 + nn) > qrow || mk_s[16 + nn] <= 0.0f) v1 = -1e30f;
      float mx = fmaxf(v0, v1);
#pragma unroll
      for (int off = 1; off < 16; off <<= 1)
        mx = fmaxf(mx, __shfl_xor(mx, off, 32));
      const float mnew = fmaxf(m_i[r], mx);
      const float e0 = __expf(v0 - mnew);
      const float e1 = __expf(v1 - mnew);
      float sum = e0 + e1;
#pragma unroll
      for (int off = 1; off < 16; off <<= 1)
        sum += __shfl_xor(sum, off, 32);
      corr[r] = __expf(m_i[r] - mnew);
      l_i[r]  = l_i[r] * corr[r] + sum;
      m_i[r]  = mnew;
      // bounce P through per-wave LDS: C-frag layout -> A-frag layout
      p_s[w][r + 8 * half][nn]      = (__bf16)e0;
      p_s[w][r + 8 * half][16 + nn] = (__bf16)e1;
    }
#pragma unroll
    for (int j = 0; j < 4; ++j)
#pragma unroll
      for (int r = 0; r < 8; ++r)
        o[j][r] *= corr[r];

    // same-wave LDS RAW fence before re-reading P as an A fragment
    asm volatile("s_wait_dscnt 0" ::: "memory");

    // ---- O += P(16x32) · V(32x64) ----
    {
      v16bf ap = load_frag_a(&p_s[w][0][0], 40);
#pragma unroll
      for (int j = 0; j < 4; ++j) {
        v16bf bv = load_frag_bT(&v_t[j * 16][0], 40);
        o[j] = wmma_bf16(ap, bv, o[j]);
      }
    }
    __syncthreads();
  }

  // ---- epilogue: normalize, store bf16 [B,T,H,HD] ----
#pragma unroll
  for (int r = 0; r < 8; ++r) {
    const int qrow   = q0 + w * 16 + r + 8 * half;
    const float inv  = 1.0f / l_i[r];
    const size_t bas = (size_t)(b * Tt + qrow) * DIM + h * HD;
#pragma unroll
    for (int j = 0; j < 4; ++j)
      Ob[bas + j * 16 + nn] = (__bf16)(o[j][r] * inv);
  }
}

// ---------------------------------------------------------------------------
// Host-side launcher. Inputs: x, sin, cos, mask, Wq, Wk, Wv, Wo (all f32).
// Workspace layout (bytes): Qf 16MB | Kf 4MB | Vf 4MB | attn(bf16) 8MB.
// ---------------------------------------------------------------------------
extern "C" void kernel_launch(void* const* d_in, const int* in_sizes, int n_in,
                              void* d_out, int out_size, void* d_ws, size_t ws_size,
                              hipStream_t stream) {
  const float* x     = (const float*)d_in[0];
  const float* sinp  = (const float*)d_in[1];
  const float* cosp  = (const float*)d_in[2];
  const float* maskp = (const float*)d_in[3];
  const float* Wq    = (const float*)d_in[4];
  const float* Wk    = (const float*)d_in[5];
  const float* Wv    = (const float*)d_in[6];
  const float* Wo    = (const float*)d_in[7];
  float* out = (float*)d_out;

  char* ws = (char*)d_ws;
  float*  Qf = (float*)ws;                                  // 4096x1024 f32
  float*  Kf = (float*)(ws + (size_t)16 * 1024 * 1024);     // 4096x256  f32
  float*  Vf = (float*)(ws + (size_t)20 * 1024 * 1024);     // 4096x256  f32
  __bf16* Ab = (__bf16*)(ws + (size_t)24 * 1024 * 1024);    // 4096x1024 bf16

  const int M = Bb * Tt;   // 4096
  dim3 blk(256);

  // Q/K/V projections (f32 A path)
  gemm_wmma_k<false><<<dim3(DIM / 64, M / 128), blk, 0, stream>>>(
      (const void*)x, Wq, Qf, M, DIM, DIM);
  gemm_wmma_k<false><<<dim3(KVD / 64, M / 128), blk, 0, stream>>>(
      (const void*)x, Wk, Kf, M, KVD, DIM);
  gemm_wmma_k<false><<<dim3(KVD / 64, M / 128), blk, 0, stream>>>(
      (const void*)x, Wv, Vf, M, KVD, DIM);

  // flash attention (RoPE + GQA + causal + padding mask)
  flash_attn_k<<<dim3(Tt / 64, HEADS, Bb), dim3(128), 0, stream>>>(
      Qf, Kf, Vf, sinp, cosp, maskp, Ab);

  // output projection (bf16 A path)
  gemm_wmma_k<true><<<dim3(DIM / 64, M / 128), blk, 0, stream>>>(
      (const void*)Ab, Wo, out, M, DIM, DIM);
}
